// MQAttention_21973052686455
// MI455X (gfx1250) — matrix operand order repaired and checked by reading, not compiled, not run on hardware
//
#include <hip/hip_runtime.h>
#include <hip/hip_bf16.h>
#include <math.h>

// MI455X / gfx1250, wave32. All matmuls via v_wmma_f32_16x16x32_f16.

typedef __attribute__((ext_vector_type(16))) _Float16 v16h;
typedef __attribute__((ext_vector_type(8)))  _Float16 v8h;
typedef __attribute__((ext_vector_type(2)))  _Float16 v2h;
typedef __attribute__((ext_vector_type(8)))  float    v8f;

#define WMMA_F16(a, b, c) \
  __builtin_amdgcn_wmma_f32_16x16x32_f16(false, (a), false, (b), (short)0, (c), false, false)

#if __has_builtin(__builtin_amdgcn_tanhf)
  #define FAST_TANH(x) __builtin_amdgcn_tanhf(x)
#elif __has_builtin(__builtin_amdgcn_tanh_f32)
  #define FAST_TANH(x) __builtin_amdgcn_tanh_f32(x)
#else
  #define FAST_TANH(x) tanhf(x)
#endif

constexpr int Bsz = 2, T = 2048, DIM = 1024, H = 16, HD = 64;
constexpr int Mrows = Bsz * T;            // 4096 token rows
constexpr int LDA_LDS = 1032;             // 1024 + 8 halves pad (row stride 2064 B, 16B-mult)

__device__ inline v16h cat8(v8h lo, v8h hi) {
  return __builtin_shufflevector(lo, hi, 0,1,2,3,4,5,6,7,8,9,10,11,12,13,14,15);
}
__device__ inline v16h load16h(const _Float16* p) {   // 32B = 2x b128, must be 16B aligned
  return cat8(*(const v8h*)p, *(const v8h*)(p + 8));
}

// ---------------------------------------------------------------------------
// Kernel 0: fp32 -> f16 weight conversion (one pass, lives in L2 afterwards).
// ---------------------------------------------------------------------------
__global__ __launch_bounds__(256)
void cvt_kernel(const float* __restrict__ src, _Float16* __restrict__ dst, int n)
{
  const int i = (blockIdx.x * 256 + threadIdx.x) * 8;
  if (i >= n) return;
  const float4 a = *(const float4*)(src + i);
  const float4 b = *(const float4*)(src + i + 4);
  v8h o;
  o[0] = (_Float16)a.x; o[1] = (_Float16)a.y; o[2] = (_Float16)a.z; o[3] = (_Float16)a.w;
  o[4] = (_Float16)b.x; o[5] = (_Float16)b.y; o[6] = (_Float16)b.z; o[7] = (_Float16)b.w;
  *(v8h*)(dst + i) = o;
}

// ---------------------------------------------------------------------------
// Kernel 1: fused QKV projection (f16 weights).  q[4096,1024], k[4096,64],
// v stored TRANSPOSED as vT[b][hd][t] (packed b128 epilogue stores).
// Block: 256 thr (8 waves) = 1 M-tile x 8 N-tiles.  grid=(256, 9) -> N=1152.
// ---------------------------------------------------------------------------
__global__ __launch_bounds__(256)
void qkv_gemm_kernel(const float* __restrict__ x,
                     const _Float16* __restrict__ wq16, const _Float16* __restrict__ wk16,
                     const _Float16* __restrict__ wv16,
                     _Float16* __restrict__ qh, _Float16* __restrict__ kh,
                     _Float16* __restrict__ vT)
{
  __shared__ alignas(16) _Float16 As[16 * LDA_LDS];
  const int tid  = threadIdx.x;
  const int row0 = blockIdx.x * 16;

  for (int idx = tid; idx < 16 * DIM; idx += 256) {
    const int r = idx >> 10, k = idx & 1023;
    As[r * LDA_LDS + k] = (_Float16)x[(row0 + r) * DIM + k];
  }
  __syncthreads();

  const int lane  = tid & 31;
  const int ntile = blockIdx.y * 8 + (tid >> 5);   // 0..71
  const int ln    = lane & 15;                     // A row in tile == B column
  const int g     = lane >> 4;                     // lane group

  const _Float16* W;
  if (ntile < 64)      W = wq16 + (size_t)( ntile       * 16) * DIM;
  else if (ntile < 68) W = wk16 + (size_t)((ntile - 64) * 16) * DIM;
  else                 W = wv16 + (size_t)((ntile - 68) * 16) * DIM;
  const _Float16* wrow = W + (size_t)ln * DIM;     // B[k][n] = W[n][k]

  v8f c = {};
  const _Float16* arow = &As[ln * LDA_LDS];
  for (int k0 = 0; k0 < DIM; k0 += 32) {
    const v16h a = cat8(*(const v8h*)(arow + k0 + g * 16),
                        *(const v8h*)(arow + k0 + 8 + g * 16));
    const _Float16* wp = wrow + k0 + g * 16;
    __builtin_prefetch(wp + 64, 0, 0);             // next-next K step
    const v16h b = load16h(wp);
    c = WMMA_F16(a, b, c);
  }

  if (ntile < 68) {                                // q / k: row-major b16 stores
    _Float16* dst; int ldd, ncol0;
    if (ntile < 64) { dst = qh; ldd = H * HD; ncol0 = ntile * 16; }
    else            { dst = kh; ldd = HD;     ncol0 = (ntile - 64) * 16; }
    #pragma unroll
    for (int r = 0; r < 8; ++r)
      dst[(size_t)(row0 + g * 8 + r) * ldd + ncol0 + ln] = (_Float16)c[r];
  } else {                                         // v: transposed, packed store
    v8h pk;
    #pragma unroll
    for (int r = 0; r < 8; ++r) pk[r] = (_Float16)c[r];
    const int bb = row0 / T, t0 = row0 - bb * T;
    const int hd = (ntile - 68) * 16 + ln;
    *(v8h*)(vT + (size_t)bb * HD * T + (size_t)hd * T + t0 + g * 8) = pk;
  }
}

// ---------------------------------------------------------------------------
// Kernel 2: RMSNorm + RoPE (interleaved pairs), in-place on f16 q / k.
// ---------------------------------------------------------------------------
__global__ __launch_bounds__(256)
void normrope_kernel(_Float16* __restrict__ qh, _Float16* __restrict__ kh,
                     const float* __restrict__ qg, const float* __restrict__ kg,
                     const float* __restrict__ fc, const float* __restrict__ fs)
{
  const int lane = threadIdx.x & 31;
  const int w    = blockIdx.x * 8 + (threadIdx.x >> 5);

  _Float16* row; const float* gam; int t;
  if (w < Bsz * T * H) { row = qh + (size_t)w * HD;  t = (w / H) % T; gam = qg; }
  else { const int w2 = w - Bsz * T * H; row = kh + (size_t)w2 * HD; t = w2 % T; gam = kg; }

  const v2h pr = *(const v2h*)(row + 2 * lane);
  const float v0 = (float)pr[0], v1 = (float)pr[1];
  float ss = v0 * v0 + v1 * v1;
  #pragma unroll
  for (int off = 16; off; off >>= 1) ss += __shfl_xor(ss, off, 32);
  const float rms = rsqrtf(ss * (1.0f / 64.0f) + 1e-6f);

  const float2 gg = *(const float2*)(gam + 2 * lane);
  const float xe = v0 * rms * gg.x;
  const float xo = v1 * rms * gg.y;
  const float c  = fc[t * (HD / 2) + lane];
  const float s  = fs[t * (HD / 2) + lane];
  v2h out; out[0] = (_Float16)(xe * c - xo * s); out[1] = (_Float16)(xe * s + xo * c);
  *(v2h*)(row + 2 * lane) = out;
}

// ---------------------------------------------------------------------------
// Kernel 3: flash attention, tanh softcap, causal, MQA (1 KV head).
// One wave per 16-row Q tile; 32 keys/step; online softmax; P transposed
// C-layout -> A-layout through wave-private LDS.  8 WMMAs / 32 keys.
// ---------------------------------------------------------------------------
__global__ __launch_bounds__(256)
void attn_kernel(const _Float16* __restrict__ qh, const _Float16* __restrict__ kh,
                 const _Float16* __restrict__ vT, _Float16* __restrict__ oh)
{
  __shared__ alignas(16) _Float16 Ps[8][16][40];   // row stride 80B (16B mult)
  const int tid = threadIdx.x;
  const int lane = tid & 31, wave = tid >> 5;
  const int g = lane >> 4, ln = lane & 15;
  const int b = blockIdx.x / H, h = blockIdx.x % H;
  const int q0 = (blockIdx.y * 8 + wave) * 16;

  // Q tile in A layout, HD=64 -> two K=32 chunks
  const _Float16* qrow = qh + ((size_t)(b * T + q0 + ln) * H + h) * HD;
  const v16h aq0 = cat8(*(const v8h*)(qrow + g * 16),      *(const v8h*)(qrow + 8 + g * 16));
  const v16h aq1 = cat8(*(const v8h*)(qrow + 32 + g * 16), *(const v8h*)(qrow + 40 + g * 16));

  float mrun[8], lrun[8], alpha[8];
  v8f acc0 = {}, acc1 = {}, acc2 = {}, acc3 = {};
  #pragma unroll
  for (int r = 0; r < 8; ++r) { mrun[r] = -INFINITY; lrun[r] = 0.0f; }

  const _Float16* kbase = kh + (size_t)(b * T) * HD;
  const _Float16* vbase = vT + (size_t)b * HD * T;

  const int nsteps = (q0 + 16 + 31) >> 5;          // causal bound
  for (int step = 0; step < nsteps; ++step) {
    const int s0 = step * 32;

    // K^T as B: columns = keys, K-dim = hd.  16 contiguous halves per lane.
    const _Float16* kr0 = kbase + (size_t)(s0 + ln) * HD;
    const _Float16* kr1 = kbase + (size_t)(s0 + 16 + ln) * HD;
    const v16h bk00 = load16h(kr0 + g * 16);
    const v16h bk01 = load16h(kr0 + 32 + g * 16);
    const v16h bk10 = load16h(kr1 + g * 16);
    const v16h bk11 = load16h(kr1 + 32 + g * 16);

    v8f s0c = {}, s1c = {};
    s0c = WMMA_F16(aq0, bk00, s0c);
    s0c = WMMA_F16(aq1, bk01, s0c);
    s1c = WMMA_F16(aq0, bk10, s1c);
    s1c = WMMA_F16(aq1, bk11, s1c);

    // softcap + causal mask + online softmax (row stats in 16-lane groups)
    #pragma unroll
    for (int r = 0; r < 8; ++r) {
      const int qrw = q0 + g * 8 + r;
      float v0 = 50.0f * FAST_TANH(s0c[r] * (0.125f * 0.02f));
      float v1 = 50.0f * FAST_TANH(s1c[r] * (0.125f * 0.02f));
      v0 += (s0 + ln      > qrw) ? -1.0e9f : 0.0f;
      v1 += (s0 + 16 + ln > qrw) ? -1.0e9f : 0.0f;
      float mnew = fmaxf(v0, v1);
      #pragma unroll
      for (int off = 8; off; off >>= 1) mnew = fmaxf(mnew, __shfl_xor(mnew, off, 32));
      const float mr = fmaxf(mrun[r], mnew);
      const float a_ = __expf(mrun[r] - mr);
      const float p0 = __expf(v0 - mr);
      const float p1 = __expf(v1 - mr);
      float ps = p0 + p1;
      #pragma unroll
      for (int off = 8; off; off >>= 1) ps += __shfl_xor(ps, off, 32);
      lrun[r] = lrun[r] * a_ + ps;
      mrun[r] = mr;
      alpha[r] = a_;
      Ps[wave][g * 8 + r][ln]      = (_Float16)p0;   // C-layout -> LDS
      Ps[wave][g * 8 + r][16 + ln] = (_Float16)p1;
    }
    #pragma unroll
    for (int r = 0; r < 8; ++r) {
      acc0[r] *= alpha[r]; acc1[r] *= alpha[r];
      acc2[r] *= alpha[r]; acc3[r] *= alpha[r];
    }

    // P back out of LDS in A layout (16x32) via b128 ds loads
    const v16h pa = cat8(*(const v8h*)&Ps[wave][ln][g * 16],
                         *(const v8h*)&Ps[wave][ln][8 + g * 16]);

    // V^T as B: column = hd, K-dim = keys (contiguous in t) -> b128 loads
    const _Float16* vk = vbase + s0 + g * 16;
    const v16h bv0 = load16h(vk + (size_t)(ln)      * T);
    const v16h bv1 = load16h(vk + (size_t)(16 + ln) * T);
    const v16h bv2 = load16h(vk + (size_t)(32 + ln) * T);
    const v16h bv3 = load16h(vk + (size_t)(48 + ln) * T);

    acc0 = WMMA_F16(pa, bv0, acc0);
    acc1 = WMMA_F16(pa, bv1, acc1);
    acc2 = WMMA_F16(pa, bv2, acc2);
    acc3 = WMMA_F16(pa, bv3, acc3);
  }

  #pragma unroll
  for (int r = 0; r < 8; ++r) {
    const int t = q0 + g * 8 + r;
    const float inv = 1.0f / lrun[r];
    _Float16* orow = oh + (size_t)(b * T + t) * (H * HD) + h * HD;
    orow[ln]      = (_Float16)(acc0[r] * inv);
    orow[16 + ln] = (_Float16)(acc1[r] * inv);
    orow[32 + ln] = (_Float16)(acc2[r] * inv);
    orow[48 + ln] = (_Float16)(acc3[r] * inv);
  }
}

// ---------------------------------------------------------------------------
// Kernel 4: output projection.  out(f32) = attn(f16,[4096,1024]) x wo16^T.
// ---------------------------------------------------------------------------
__global__ __launch_bounds__(256)
void out_gemm_kernel(const _Float16* __restrict__ oh, const _Float16* __restrict__ wo16,
                     float* __restrict__ out)
{
  __shared__ alignas(16) _Float16 As[16 * LDA_LDS];
  const int tid  = threadIdx.x;
  const int row0 = blockIdx.x * 16;

  for (int idx = tid; idx < 16 * DIM; idx += 256) {
    const int r = idx >> 10, k = idx & 1023;
    As[r * LDA_LDS + k] = oh[(size_t)(row0 + r) * DIM + k];
  }
  __syncthreads();

  const int lane  = tid & 31;
  const int ntile = blockIdx.y * 8 + (tid >> 5);   // 0..63
  const int ln    = lane & 15;
  const int g     = lane >> 4;
  const _Float16* wrow = wo16 + (size_t)(ntile * 16 + ln) * DIM;

  v8f c = {};
  const _Float16* arow = &As[ln * LDA_LDS];
  for (int k0 = 0; k0 < DIM; k0 += 32) {
    const v16h a = cat8(*(const v8h*)(arow + k0 + g * 16),
                        *(const v8h*)(arow + k0 + 8 + g * 16));
    const _Float16* wp = wrow + k0 + g * 16;
    __builtin_prefetch(wp + 64, 0, 0);
    const v16h b = load16h(wp);
    c = WMMA_F16(a, b, c);
  }
  #pragma unroll
  for (int r = 0; r < 8; ++r)
    out[(size_t)(row0 + g * 8 + r) * DIM + ntile * 16 + ln] = c[r];
}

// ---------------------------------------------------------------------------
extern "C" void kernel_launch(void* const* d_in, const int* in_sizes, int n_in,
                              void* d_out, int out_size, void* d_ws, size_t ws_size,
                              hipStream_t stream) {
  const float* x  = (const float*)d_in[0];
  const float* wq = (const float*)d_in[1];
  const float* wk = (const float*)d_in[2];
  const float* wv = (const float*)d_in[3];
  const float* wo = (const float*)d_in[4];
  const float* qg = (const float*)d_in[5];
  const float* kg = (const float*)d_in[6];
  const float* fc = (const float*)d_in[7];
  const float* fs = (const float*)d_in[8];
  // d_in[9] = mask: applied analytically (causal -1e9 add), not loaded.
  float* out = (float*)d_out;

  char* ws = (char*)d_ws;
  const size_t MiB = 1024 * 1024;
  _Float16* qh   = (_Float16*)(ws);                       // 8 MiB  [b][t][h][hd]
  _Float16* kh   = (_Float16*)(ws + 8 * MiB);             // 512 KiB [b][t][hd]
  _Float16* vT   = (_Float16*)(ws + 8 * MiB + 512 * 1024);// 512 KiB [b][hd][t]
  _Float16* oh   = (_Float16*)(ws + 9 * MiB);             // 8 MiB  [b][t][h*hd]
  _Float16* wq16 = (_Float16*)(ws + 17 * MiB);            // 2 MiB
  _Float16* wk16 = (_Float16*)(ws + 19 * MiB);            // 128 KiB
  _Float16* wv16 = (_Float16*)(ws + 19 * MiB + 128 * 1024);// 128 KiB
  _Float16* wo16 = (_Float16*)(ws + 20 * MiB);            // 2 MiB

  const dim3 blk(256);
  cvt_kernel<<<dim3((DIM * DIM) / 2048), blk, 0, stream>>>(wq, wq16, DIM * DIM);
  cvt_kernel<<<dim3((HD * DIM) / 2048),  blk, 0, stream>>>(wk, wk16, HD * DIM);
  cvt_kernel<<<dim3((HD * DIM) / 2048),  blk, 0, stream>>>(wv, wv16, HD * DIM);
  cvt_kernel<<<dim3((DIM * DIM) / 2048), blk, 0, stream>>>(wo, wo16, DIM * DIM);

  qkv_gemm_kernel<<<dim3(Mrows / 16, 9), blk, 0, stream>>>(x, wq16, wk16, wv16, qh, kh, vT);
  normrope_kernel<<<dim3((Bsz * T * H + Bsz * T) / 8), blk, 0, stream>>>(qh, kh, qg, kg, fc, fs);
  attn_kernel<<<dim3(Bsz * H, T / 128), blk, 0, stream>>>(qh, kh, vT, oh);
  out_gemm_kernel<<<dim3(Mrows / 16, 8), blk, 0, stream>>>(oh, wo16, out);
}
